// SemanticComposite_2508260901178
// MI455X (gfx1250) — matrix operand order repaired and checked by reading, not compiled, not run on hardware
//
#include <hip/hip_runtime.h>
#include <cstdint>
#include <cstddef>

typedef __bf16 bf16;
typedef __attribute__((ext_vector_type(16))) __bf16 v16bf;
typedef __attribute__((ext_vector_type(8)))  __bf16 v8bf;
typedef __attribute__((ext_vector_type(8)))  float  v8f;

#define BB 4
#define LL 2048
#define DD 768
#define KD 1536   // 2*D

// ---- optional gfx1250 async global->LDS path -------------------------------

#if __has_builtin(__builtin_amdgcn_global_load_async_to_lds_b128) && \
    __has_builtin(__builtin_amdgcn_s_wait_asynccnt)
#define USE_ASYNC_LDS 1
typedef int v4i __attribute__((vector_size(16)));
typedef __attribute__((address_space(1))) v4i gv4i_t;   // global int4
typedef __attribute__((address_space(3))) v4i lv4i_t;   // LDS int4
static __device__ __forceinline__ void async_cp16(const void* g, void* l) {
  __builtin_amdgcn_global_load_async_to_lds_b128((gv4i_t*)g, (lv4i_t*)l, 0, 0);
}
static __device__ __forceinline__ void async_wait0() {
  __builtin_amdgcn_s_wait_asynccnt(0);
}
#else
#define USE_ASYNC_LDS 0
#endif

// ---- helpers ---------------------------------------------------------------

static __device__ __forceinline__ bf16 f2bf(float f) {
  union { float f; uint32_t u; } c; c.f = f;
  uint32_t u = c.u;
  uint32_t r = (u + 0x7FFFu + ((u >> 16) & 1u)) >> 16;   // round-to-nearest-even
  union { unsigned short s; bf16 b; } o; o.s = (unsigned short)r;
  return o.b;
}

static __device__ __forceinline__ v16bf joinv(v8bf lo, v8bf hi) {
  v16bf r;
#pragma unroll
  for (int i = 0; i < 8; ++i) { r[i] = lo[i]; r[i + 8] = hi[i]; }
  return r;
}

static __device__ __forceinline__ v8f wmma_bf16(v16bf a, v16bf b, v8f c) {
  return __builtin_amdgcn_wmma_f32_16x16x32_bf16(
      /*neg_a=*/false, a, /*neg_b=*/false, b,
      /*c_mod=*/(short)0, c, /*reuse_a=*/false, /*reuse_b=*/false);
}

static __device__ __forceinline__ float sigmoidf_(float v) {
  return 1.0f / (1.0f + __expf(-v));
}

// ---- kernel 0: convert gate weights to bf16 --------------------------------

__global__ void k_convw(const float* __restrict__ zw, const float* __restrict__ rw,
                        const float* __restrict__ fw,
                        bf16* __restrict__ zwb, bf16* __restrict__ rwb,
                        bf16* __restrict__ fwb, int n) {
  int i = blockIdx.x * blockDim.x + threadIdx.x;
  if (i < n) {
    zwb[i] = f2bf(zw[i]);
    rwb[i] = f2bf(rw[i]);
    fwb[i] = f2bf(fw[i]);
  }
}

// ---- kernel 1: per-row prep: a=x.w1, b=x.w2, bf16 copies -------------------

__global__ void __launch_bounds__(256)
k_prep(const float* __restrict__ x, const float* __restrict__ attw,
       bf16* __restrict__ xbf, bf16* __restrict__ xw3,
       float* __restrict__ av, float* __restrict__ bv) {
  __shared__ float r1[256];
  __shared__ float r2[256];
  const int row = blockIdx.x;          // b*L + l
  const float* xr = x + (size_t)row * DD;
  float s1 = 0.f, s2 = 0.f;
#pragma unroll
  for (int k = 0; k < 3; ++k) {        // 768 = 3 * 256
    int d = threadIdx.x + k * 256;
    float v = xr[d];
    s1 += v * attw[d];
    s2 += v * attw[DD + d];
    xbf[(size_t)row * DD + d] = f2bf(v);
    xw3[(size_t)row * DD + d] = f2bf(v * attw[2 * DD + d]);
  }
  r1[threadIdx.x] = s1; r2[threadIdx.x] = s2;
  __syncthreads();
  for (int off = 128; off > 0; off >>= 1) {
    if (threadIdx.x < (unsigned)off) {
      r1[threadIdx.x] += r1[threadIdx.x + off];
      r2[threadIdx.x] += r2[threadIdx.x + off];
    }
    __syncthreads();
  }
  if (threadIdx.x == 0) { av[row] = r1[0]; bv[row] = r2[0]; }
}

// ---- kernel 1b: coalesced 64x64 tile transpose x -> xT (bf16) --------------
// grid = (L/64, D/64, B), block = 256.

__global__ void __launch_bounds__(256)
k_tr(const float* __restrict__ x, bf16* __restrict__ xT) {
  __shared__ bf16 tile[64][68];        // padded rows vs bank conflicts
  const int bt = blockIdx.z;
  const int l0 = blockIdx.x << 6;
  const int d0 = blockIdx.y << 6;
  const int tid = threadIdx.x;
#pragma unroll
  for (int k = 0; k < 16; ++k) {
    int idx = tid + k * 256;
    int r = idx >> 6, c = idx & 63;
    tile[r][c] = f2bf(x[((size_t)(bt * LL + l0 + r)) * DD + d0 + c]);
  }
  __syncthreads();
#pragma unroll
  for (int k = 0; k < 16; ++k) {
    int idx = tid + k * 256;
    int dr = idx >> 6, lc = idx & 63;
    xT[((size_t)(bt * DD + d0 + dr)) * LL + l0 + lc] = tile[lc][dr];
  }
}

// ---- kernel 2: flash-style scores + softmax + attn, all via WMMA -----------
// grid = B * (L/16); block = 256 (8 wave32).  Per block: one 16-row i-tile.

__global__ void __launch_bounds__(256)
k_attn(const bf16* __restrict__ xbf, const bf16* __restrict__ xw3,
       const bf16* __restrict__ xT, const float* __restrict__ av,
       const float* __restrict__ bv, bf16* __restrict__ attnbf) {
  __shared__ bf16  sA[16][DD];      // (x*w3) i-tile, A operand for scores (24 KB)
  __shared__ float sS[16][128];     // score chunk (8 KB)
  __shared__ bf16  sP[16][128];     // prob chunk bf16 (4 KB)
  __shared__ float sRed[16][16];    // softmax reduction scratch
  __shared__ float m_s[16], l_s[16], scale_s[16], arow[16];

  const int tid  = threadIdx.x;
  const int lane = tid & 31;
  const int wv   = tid >> 5;          // 0..7
  const int bt   = blockIdx.x >> 7;   // / (L/16)
  const int it   = blockIdx.x & 127;
  const int i0   = it << 4;
  const int n    = lane & 15;
  const int hi   = (lane < 16) ? 0 : 1;

  // stage A (x*w3 rows i0..i0+15) into LDS: a flat 24 KB contiguous block
  {
    const bf16* srcA = xw3 + ((size_t)(bt * LL + i0)) * DD;
    bf16* dstA = &sA[0][0];
#if USE_ASYNC_LDS
#pragma unroll
    for (int k = 0; k < 6; ++k) {            // 16*768 / (256*8)
      int off = (tid + k * 256) * 8;         // 16B granules
      async_cp16(srcA + off, dstA + off);
    }
#else
    for (int idx = tid; idx < 16 * DD; idx += 256) dstA[idx] = srcA[idx];
#endif
  }
  if (tid < 16) {
    m_s[tid] = -3.0e38f;
    l_s[tid] = 0.f;
    arow[tid] = av[bt * LL + i0 + tid];
  }
#if USE_ASYNC_LDS
  async_wait0();
#endif
  __syncthreads();

  v8f O[6];
  {
    v8f zero = {};
#pragma unroll
    for (int t = 0; t < 6; ++t) O[t] = zero;
  }

  const int srow = tid >> 4;       // softmax: row handled by this thread
  const int sseg = tid & 15;       // softmax: 8-col segment within the row

  for (int jbase = 0; jbase < LL; jbase += 128) {
    // ---- phase A: each wave computes one 16x16 score tile -------------------
    const int j = jbase + wv * 16 + n;               // B column owned by lane
    const bf16* xrow = xbf + ((size_t)(bt * LL + j)) * DD + hi * 16;
    if (jbase + 128 < LL)
      __builtin_prefetch(xbf + ((size_t)(bt * LL + j + 128)) * DD, 0, 3);
    v8f s = {};
#pragma unroll 4
    for (int k0 = 0; k0 < DD; k0 += 32) {
      int aoff = k0 + hi * 16;
      v16bf afrag = joinv(*(const v8bf*)&sA[n][aoff], *(const v8bf*)&sA[n][aoff + 8]);
      v16bf bfrag = joinv(*(const v8bf*)(xrow + k0), *(const v8bf*)(xrow + k0 + 8));
      s = wmma_bf16(afrag, bfrag, s);
    }
    float bj = bv[bt * LL + j];
#pragma unroll
    for (int v = 0; v < 8; ++v) {
      int m = v + hi * 8;
      sS[m][wv * 16 + n] = s[v] + arow[m] + bj;
    }
    __syncthreads();

    // ---- phase B: parallel online-softmax update (all 256 threads) ----------
    {   // step 1: per-segment max (8 cols each)
      float mx = -3.0e38f;
#pragma unroll
      for (int c = 0; c < 8; ++c) mx = fmaxf(mx, sS[srow][sseg * 8 + c]);
      sRed[srow][sseg] = mx;
    }
    __syncthreads();
    if (tid < 16) {  // step 2: row max, rescale factor
      float mx = -3.0e38f;
#pragma unroll
      for (int c = 0; c < 16; ++c) mx = fmaxf(mx, sRed[tid][c]);
      float newm = fmaxf(m_s[tid], mx);
      scale_s[tid] = __expf(m_s[tid] - newm);
      m_s[tid] = newm;
    }
    __syncthreads();
    {   // step 3: exponentiate, write bf16 probs, partial sums
      float newm = m_s[srow];
      float sum = 0.f;
#pragma unroll
      for (int c = 0; c < 8; ++c) {
        float p = __expf(sS[srow][sseg * 8 + c] - newm);
        sum += p;
        sP[srow][sseg * 8 + c] = f2bf(p);
      }
      sRed[srow][sseg] = sum;
    }
    __syncthreads();
    if (tid < 16) {  // step 4: fold partial sums into running denominator
      float sum = 0.f;
#pragma unroll
      for (int c = 0; c < 16; ++c) sum += sRed[tid][c];
      l_s[tid] = l_s[tid] * scale_s[tid] + sum;
    }

    // ---- phase C: rescale O, then O += P_chunk (16x128) @ x_chunk (128x96) --
    float sc[8];
#pragma unroll
    for (int v = 0; v < 8; ++v) sc[v] = scale_s[v + hi * 8];
#pragma unroll
    for (int t = 0; t < 6; ++t)
#pragma unroll
      for (int v = 0; v < 8; ++v) O[t][v] *= sc[v];

#pragma unroll
    for (int k0 = 0; k0 < 128; k0 += 32) {
      int aoff = k0 + hi * 16;
      v16bf afrag = joinv(*(const v8bf*)&sP[n][aoff], *(const v8bf*)&sP[n][aoff + 8]);
#pragma unroll
      for (int t = 0; t < 6; ++t) {
        int d = wv * 96 + t * 16 + n;                // B column = feature d
        const bf16* bp = xT + ((size_t)(bt * DD + d)) * LL + jbase + k0 + hi * 16;
        v16bf bfrag = joinv(*(const v8bf*)bp, *(const v8bf*)(bp + 8));
        O[t] = wmma_bf16(afrag, bfrag, O[t]);
      }
    }
    __syncthreads();
  }

  // ---- finalize: divide by row sums, store attn as bf16 ----------------------
  float linv[8];
#pragma unroll
  for (int v = 0; v < 8; ++v) linv[v] = 1.0f / l_s[v + hi * 8];
#pragma unroll
  for (int t = 0; t < 6; ++t) {
    int d = wv * 96 + t * 16 + n;
#pragma unroll
    for (int v = 0; v < 8; ++v) {
      int m = v + hi * 8;
      attnbf[((size_t)(bt * LL + i0 + m)) * DD + d] = f2bf(O[t][v] * linv[v]);
    }
  }
}

// ---- kernel 3: fused gate GEMMs z/r/f + epilogue ---------------------------
// grid = (B*L)/16; block = 512 (16 wave32). Each wave owns 48 output columns.

__global__ void __launch_bounds__(512)
k_gates(const float* __restrict__ x, const bf16* __restrict__ attnbf,
        const bf16* __restrict__ zwb, const bf16* __restrict__ rwb,
        const bf16* __restrict__ fwb,
        const float* __restrict__ zb, const float* __restrict__ rb,
        const float* __restrict__ fb, float* __restrict__ out) {
  __shared__ bf16 sXA[16][KD];      // xa = [x | attn] M-tile, 48 KB

  const int tid  = threadIdx.x;
  const int lane = tid & 31;
  const int wv   = tid >> 5;        // 0..15
  const int g0   = blockIdx.x << 4; // row base in [0, B*L)
  const int n    = lane & 15;
  const int hi   = (lane < 16) ? 0 : 1;

#if USE_ASYNC_LDS
  // attn half: async bf16 global->LDS copies (issue first, overlap with cvt)
#pragma unroll
  for (int k = 0; k < 3; ++k) {
    int ch = tid + k * 512;          // 1536 x 16B granules
    int r = ch / 96, wi = ch % 96;
    async_cp16(attnbf + (size_t)(g0 + r) * DD + wi * 8, &sXA[r][DD] + wi * 8);
  }
#endif
  // x half: f32 -> bf16 conversion
  for (int idx = tid; idx < 16 * DD; idx += 512) {
    int r = idx / DD, c = idx % DD;
    sXA[r][c] = f2bf(x[(size_t)(g0 + r) * DD + c]);
  }
#if USE_ASYNC_LDS
  async_wait0();
#else
  for (int idx = tid; idx < 16 * DD; idx += 512) {
    int r = idx / DD, c = idx % DD;
    sXA[r][DD + c] = attnbf[(size_t)(g0 + r) * DD + c];
  }
#endif
  __syncthreads();

  v8f az[3], ag[3], af[3];
  {
    v8f zero = {};
#pragma unroll
    for (int t = 0; t < 3; ++t) { az[t] = zero; ag[t] = zero; af[t] = zero; }
  }

  const int klosel = hi * 16;
  for (int k0 = 0; k0 < KD; k0 += 32) {
    int aoff = k0 + hi * 16;
    v16bf afrag = joinv(*(const v8bf*)&sXA[n][aoff], *(const v8bf*)&sXA[n][aoff + 8]);
#pragma unroll
    for (int t = 0; t < 3; ++t) {
      int col = wv * 48 + t * 16 + n;                // output channel owned by lane
      const bf16* pz = zwb + (size_t)col * KD + k0 + klosel;
      const bf16* pr = rwb + (size_t)col * KD + k0 + klosel;
      const bf16* pf = fwb + (size_t)col * KD + k0 + klosel;
      az[t] = wmma_bf16(afrag, joinv(*(const v8bf*)pz, *(const v8bf*)(pz + 8)), az[t]);
      ag[t] = wmma_bf16(afrag, joinv(*(const v8bf*)pr, *(const v8bf*)(pr + 8)), ag[t]);
      af[t] = wmma_bf16(afrag, joinv(*(const v8bf*)pf, *(const v8bf*)(pf + 8)), af[t]);
    }
  }

#pragma unroll
  for (int t = 0; t < 3; ++t) {
    int col = wv * 48 + t * 16 + n;
    float zbias = zb[col], rbias = rb[col], fbias = fb[col];
#pragma unroll
    for (int v = 0; v < 8; ++v) {
      int row = g0 + v + hi * 8;
      float zv = tanhf(az[t][v] + zbias);
      float rv = sigmoidf_(ag[t][v] + rbias);
      float fv = sigmoidf_(af[t][v] + fbias);
      float xv = x[(size_t)row * DD + col];
      out[(size_t)row * DD + col] = rv * xv + fv * zv;
    }
  }
}

// ---- launcher --------------------------------------------------------------

extern "C" void kernel_launch(void* const* d_in, const int* in_sizes, int n_in,
                              void* d_out, int out_size, void* d_ws, size_t ws_size,
                              hipStream_t stream) {
  (void)in_sizes; (void)n_in; (void)out_size; (void)ws_size;
  const float* x    = (const float*)d_in[0];
  const float* attw = (const float*)d_in[1];
  const float* zw   = (const float*)d_in[2];
  const float* zb   = (const float*)d_in[3];
  const float* rw   = (const float*)d_in[4];
  const float* rb   = (const float*)d_in[5];
  const float* fw   = (const float*)d_in[6];
  const float* fb   = (const float*)d_in[7];
  float* out = (float*)d_out;

  char* w = (char*)d_ws;
  const size_t nx = (size_t)BB * LL * DD;
  bf16* xbf    = (bf16*)w; w += nx * 2;
  bf16* xw3    = (bf16*)w; w += nx * 2;
  bf16* xT     = (bf16*)w; w += nx * 2;
  bf16* attnbf = (bf16*)w; w += nx * 2;
  const size_t nw = (size_t)DD * KD;
  bf16* zwb = (bf16*)w; w += nw * 2;
  bf16* rwb = (bf16*)w; w += nw * 2;
  bf16* fwb = (bf16*)w; w += nw * 2;
  float* av = (float*)w; w += (size_t)BB * LL * 4;
  float* bv = (float*)w; w += (size_t)BB * LL * 4;

  k_convw<<<(int)((nw + 255) / 256), 256, 0, stream>>>(zw, rw, fw, zwb, rwb, fwb, (int)nw);
  k_prep <<<BB * LL, 256, 0, stream>>>(x, attw, xbf, xw3, av, bv);
  dim3 trg(LL / 64, DD / 64, BB);
  k_tr   <<<trg, 256, 0, stream>>>(x, xT);
  k_attn <<<BB * (LL / 16), 256, 0, stream>>>(xbf, xw3, xT, av, bv, attnbf);
  k_gates<<<(BB * LL) / 16, 512, 0, stream>>>(x, attnbf, zwb, rwb, fwb, zb, rb, fb, out);
}
